// MPNEncoder_44263932952953
// MI455X (gfx1250) — matrix operand order repaired and checked by reading, not compiled, not run on hardware
//
#include <hip/hip_runtime.h>
#include <hip/hip_bf16.h>
#include <math.h>

typedef __bf16 bf16;
typedef __attribute__((ext_vector_type(16))) __bf16 v16bf;
typedef __attribute__((ext_vector_type(8)))  float  v8f;

union Frag32B { v16bf v; uint4 u[2]; };

// ---------------------------------------------------------------------------
// WMMA GEMM: C[M,N] = act( A[M,Kp](bf16) * W[Npad,Kp](bf16)^T + addend + bias )
// 256 threads = 8 waves; wave w computes rows [m0+32w, +32), cols [n0, n0+64).
// The 64 x Ktot weight panel for this block is staged in LDS once. In the
// K-loop both operand streams are software-pipelined:
//   - A fragments (global): loads for k0+32 issued before the wmma on k0
//   - B fragments (LDS): rotating 2-deep buffer; ds loads for fragment j+1
//     (wrapping into next K-step) issued before the wmma pair on fragment j,
//     so the wait ahead of each wmma pair is dscnt<=2, not dscnt==0.
// Fragment layouts per ISA 7.12.2.
// ---------------------------------------------------------------------------
__global__ __launch_bounds__(256)
void gemm_bf16_wmma(const bf16* __restrict__ Amat, int lda, int Mtot,
                    const bf16* __restrict__ Wmat, int ldb,
                    const float* __restrict__ addend, int ldadd,
                    const float* __restrict__ bias,
                    float* __restrict__ Cout, int ldc,
                    int Ncols, int Ktot, int do_relu)
{
  extern __shared__ bf16 lds[];                 // [64][Ktot + 8]
  const int LDSK = Ktot + 8;                    // padded row stride (halves)
  const int lane = threadIdx.x & 31;
  const int wave = threadIdx.x >> 5;
  const int m0 = blockIdx.x * 256 + wave * 32;
  const int n0 = blockIdx.y * 64;

  // ---- cooperative stage of the 64-row weight panel into LDS ----
  {
    const int k8 = Ktot >> 3;                   // uint4 (8-half) chunks per row
    for (int idx = threadIdx.x; idx < 64 * k8; idx += 256) {
      const int r  = idx / k8;
      const int kc = (idx - r * k8) << 3;
      uint4 v = *(const uint4*)(Wmat + (size_t)(n0 + r) * ldb + kc);
      *(uint4*)(lds + (size_t)r * LDSK + kc) = v;
    }
  }
  __syncthreads();

  // ---- A-fragment pointers (two 16-row subtiles, rows clamped) ----
  int r0 = m0 + (lane & 15);       if (r0 >= Mtot) r0 = Mtot - 1;
  int r1 = m0 + 16 + (lane & 15);  if (r1 >= Mtot) r1 = Mtot - 1;
  const int kbA = (lane & 16) ? 8 : 0;
  const bf16* aptr0 = Amat + (size_t)r0 * lda + kbA;
  const bf16* aptr1 = Amat + (size_t)r1 * lda + kbA;
  const int kbB = (lane & 16) ? 8 : 0;
  const bf16* bl = lds + (size_t)(lane & 15) * LDSK + kbB;

  v8f acc[2][4] = {};

  // preload k0 = 0 A fragments and the first B fragment (j = 0)
  Frag32B afc[2];
  afc[0].u[0] = *(const uint4*)(aptr0);
  afc[0].u[1] = *(const uint4*)(aptr0 + 16);
  afc[1].u[0] = *(const uint4*)(aptr1);
  afc[1].u[1] = *(const uint4*)(aptr1 + 16);
  Frag32B bcur;
  bcur.u[0] = *(const uint4*)(bl);
  bcur.u[1] = *(const uint4*)(bl + 16);

  for (int k0 = 0; k0 < Ktot; k0 += 32) {
    // pipeline: issue next-iteration A loads before consuming current frags
    const int kn = (k0 + 32 < Ktot) ? (k0 + 32) : k0;
    Frag32B afn[2];
    afn[0].u[0] = *(const uint4*)(aptr0 + kn);
    afn[0].u[1] = *(const uint4*)(aptr0 + kn + 16);
    afn[1].u[0] = *(const uint4*)(aptr1 + kn);
    afn[1].u[1] = *(const uint4*)(aptr1 + kn + 16);
    __builtin_prefetch(aptr0 + k0 + 256, 0, 1);  // global_prefetch_b8
    __builtin_prefetch(aptr1 + k0 + 256, 0, 1);

#pragma unroll
    for (int j = 0; j < 4; ++j) {
      // rotate: load fragment j+1 (wrapping to next K-step's j=0) first
      const int jn = (j + 1) & 3;
      const int kb = (j == 3) ? kn : k0;
      const bf16* bp = bl + (size_t)(16 * jn) * LDSK + kb;
      Frag32B bnext;
      bnext.u[0] = *(const uint4*)(bp);
      bnext.u[1] = *(const uint4*)(bp + 16);
      acc[0][j] = __builtin_amdgcn_wmma_f32_16x16x32_bf16(
          false, afc[0].v, false, bcur.v, (short)0, acc[0][j], false, false);
      acc[1][j] = __builtin_amdgcn_wmma_f32_16x16x32_bf16(
          false, afc[1].v, false, bcur.v, (short)0, acc[1][j], false, false);
      bcur = bnext;
    }
    afc[0] = afn[0];
    afc[1] = afn[1];
  }

  // ---- epilogue: elem r -> row sub*16 + r + 8*(lane>=16), col n0+16j+(lane&15)
  const int cloc = lane & 15;
#pragma unroll
  for (int i = 0; i < 2; ++i) {
    const int rbase = m0 + i * 16 + ((lane >> 4) << 3);
#pragma unroll
    for (int j = 0; j < 4; ++j) {
      const int col = n0 + 16 * j + cloc;
      if (col >= Ncols) continue;
      const float bval = bias ? bias[col] : 0.0f;
#pragma unroll
      for (int r = 0; r < 8; ++r) {
        const int row = rbase + r;
        if (row >= Mtot) continue;
        float v = acc[i][j][r] + bval;
        if (addend) v += addend[(size_t)row * ldadd + col];
        if (do_relu) v = v > 0.0f ? v : 0.0f;
        Cout[(size_t)row * ldc + col] = v;
      }
    }
  }
}

// ------------------------- f32 -> padded bf16 conversion -------------------
__global__ void cvt_pad_bf16(const float* __restrict__ in, int M, int K,
                             bf16* __restrict__ out, int Mo, int Ko)
{
  size_t i = (size_t)blockIdx.x * blockDim.x + threadIdx.x;
  size_t tot = (size_t)Mo * Ko;
  if (i >= tot) return;
  int r = (int)(i / Ko), c = (int)(i % Ko);
  float v = (r < M && c < K) ? in[(size_t)r * K + c] : 0.0f;
  out[i] = (bf16)v;
}

// ---------------- neighbor gather: message_atom += sum*max -----------------
__global__ __launch_bounds__(256)
void atom_agg_k(const float* __restrict__ msg_bond, const int* __restrict__ a2b,
                float* __restrict__ msg_atom, int H)
{
  __shared__ int nb[6];
  const int a = blockIdx.x;
  if (threadIdx.x < 6) nb[threadIdx.x] = a2b[(size_t)a * 6 + threadIdx.x];
  __syncthreads();
  for (int h = threadIdx.x; h < H; h += blockDim.x) {
    float s = 0.0f, m = 0.0f;
#pragma unroll
    for (int j = 0; j < 6; ++j) {
      float v = msg_bond[(size_t)nb[j] * H + h];
      s += v;
      m = (j == 0) ? v : fmaxf(m, v);
    }
    msg_atom[(size_t)a * H + h] += s * m;
  }
}

// ---- bond message: tmp = msg_atom[b2a] - msg_bond[b2revb]  (bf16, padded) --
__global__ __launch_bounds__(256)
void bond_tmp_k(const float* __restrict__ msg_atom, const float* __restrict__ msg_bond,
                const int* __restrict__ b2a, const int* __restrict__ b2revb,
                bf16* __restrict__ out, int H, int Ho)
{
  const int b = blockIdx.x;
  const int a  = b2a[b];
  const int rb = b2revb[b];
  for (int h = threadIdx.x; h < Ho; h += blockDim.x) {
    float v = 0.0f;
    if (h < H) v = msg_atom[(size_t)a * H + h] - msg_bond[(size_t)rb * H + h];
    out[(size_t)b * Ho + h] = (bf16)v;
  }
}

// -------- final concat row: [s*m | message_atom | input_atom | 0pad] -------
__global__ __launch_bounds__(256)
void final_concat_k(const float* __restrict__ msg_bond, const int* __restrict__ a2b,
                    const float* __restrict__ msg_atom, const float* __restrict__ input_atom,
                    bf16* __restrict__ out, int H, int Ko)
{
  __shared__ int nb[6];
  const int a = blockIdx.x;
  if (threadIdx.x < 6) nb[threadIdx.x] = a2b[(size_t)a * 6 + threadIdx.x];
  __syncthreads();
  for (int h = threadIdx.x; h < Ko; h += blockDim.x) {
    float v = 0.0f;
    if (h < H) {
      float s = 0.0f, m = 0.0f;
#pragma unroll
      for (int j = 0; j < 6; ++j) {
        float x = msg_bond[(size_t)nb[j] * H + h];
        s += x;
        m = (j == 0) ? x : fmaxf(m, x);
      }
      v = s * m;
    } else if (h < 2 * H) {
      v = msg_atom[(size_t)a * H + (h - H)];
    } else if (h < 3 * H) {
      v = input_atom[(size_t)a * H + (h - 2 * H)];
    }
    out[(size_t)a * Ko + h] = (bf16)v;
  }
}

// ---------------- x = relu(node[1:] + gru_bias)  (bf16 padded) -------------
__global__ __launch_bounds__(256)
void msg_x_k(const float* __restrict__ node, const float* __restrict__ gbias,
             bf16* __restrict__ x, int H, int Ho)
{
  const int r = blockIdx.x;                  // 0 .. NT-1 ; atom = r+1
  const float* nrow = node + (size_t)(r + 1) * H;
  for (int h = threadIdx.x; h < Ho; h += blockDim.x) {
    float v = 0.0f;
    if (h < H) { v = nrow[h] + gbias[h]; v = v > 0.0f ? v : 0.0f; }
    x[(size_t)r * Ho + h] = (bf16)v;
  }
}

// -------- h0 = per-molecule max over pre-relu node (both directions) -------
__global__ __launch_bounds__(256)
void h0_k(const float* __restrict__ node, float* __restrict__ hf0, float* __restrict__ hf1,
          bf16* __restrict__ hb0, bf16* __restrict__ hb1, int H, int Ho, int T)
{
  const int mol = blockIdx.x;
  for (int h = threadIdx.x; h < Ho; h += blockDim.x) {
    float v = 0.0f;
    if (h < H) {
      const float* p = node + ((size_t)mol * T + 1) * H + h;
      v = p[0];
      for (int t = 1; t < T; ++t) v = fmaxf(v, p[(size_t)t * H]);
      hf0[(size_t)mol * H + h] = v;
      hf1[(size_t)mol * H + h] = v;
    }
    bf16 bv = (bf16)(h < H ? v : 0.0f);
    hb0[(size_t)mol * Ho + h] = bv;
    hb1[(size_t)mol * Ho + h] = bv;
  }
}

// ------------------------------ GRU gate step ------------------------------
__global__ __launch_bounds__(256)
void gru_gate_k(const float* __restrict__ gi, const float* __restrict__ gh,
                float* __restrict__ hf, bf16* __restrict__ hb,
                bf16* __restrict__ full, int t_eff, int colbase,
                int H, int Ho, int ldfull, int T)
{
  const int mol = blockIdx.x;
  const size_t girow = ((size_t)mol * T + t_eff) * (size_t)(3 * H);
  const size_t ghrow = (size_t)mol * (3 * H);
  for (int h = threadIdx.x; h < H; h += blockDim.x) {
    float ir = gi[girow + h];
    float iz = gi[girow + H + h];
    float in = gi[girow + 2 * H + h];
    float hr = gh[ghrow + h];
    float hz = gh[ghrow + H + h];
    float hn = gh[ghrow + 2 * H + h];
    float r = 1.0f / (1.0f + __expf(-(ir + hr)));
    float z = 1.0f / (1.0f + __expf(-(iz + hz)));
    float n = tanhf(in + r * hn);
    float hp = hf[(size_t)mol * H + h];
    float hnew = (1.0f - z) * n + z * hp;
    hf[(size_t)mol * H + h] = hnew;
    hb[(size_t)mol * Ho + h] = (bf16)hnew;
    full[((size_t)mol * T + t_eff) * ldfull + colbase + h] = (bf16)hnew;
  }
}

// ------------------- zero the K-pad columns of `full` ----------------------
__global__ void fill_pad_k(bf16* __restrict__ full, int rows, int ld, int c0, int c1)
{
  size_t i = (size_t)blockIdx.x * blockDim.x + threadIdx.x;
  const int w = c1 - c0;
  if (i >= (size_t)rows * w) return;
  int r = (int)(i / w), c = (int)(i % w);
  full[(size_t)r * ld + c0 + c] = (bf16)0.0f;
}

// ===========================================================================
extern "C" void kernel_launch(void* const* d_in, const int* in_sizes, int n_in,
                              void* d_out, int out_size, void* d_ws, size_t ws_size,
                              hipStream_t stream)
{
  const float* f_atoms  = (const float*)d_in[0];
  const float* f_bonds  = (const float*)d_in[1];
  const float* Wi_atom  = (const float*)d_in[2];
  const float* Wi_bond  = (const float*)d_in[3];
  const float* Wh       = (const float*)d_in[4];
  const float* W_lr     = (const float*)d_in[5];
  const float* Wo_w     = (const float*)d_in[6];
  const float* Wo_b     = (const float*)d_in[7];
  const float* gru_bias = (const float*)d_in[8];
  const float* gru_wih  = (const float*)d_in[9];
  const float* gru_whh  = (const float*)d_in[10];
  const float* gru_bih  = (const float*)d_in[11];
  const float* gru_bhh  = (const float*)d_in[12];
  const int*   a2b      = (const int*)d_in[13];
  const int*   b2a      = (const int*)d_in[14];
  const int*   b2revb   = (const int*)d_in[15];

  constexpr int H = 300, Hp = 320;      // hidden, K-pad (x32)
  constexpr int T = 50, NM = 2048, NT = NM * T;     // 102400
  constexpr int A_ = NT + 1;            // 102401 atoms
  constexpr int B_ = 2 * NT + 1;        // 204801 bonds
  constexpr int AF = 133, BF = 147, Fp = 160;
  constexpr int H3 = 900, H3p = 928;    // concat K pad (x32)
  constexpr int H2 = 600, H2p = 640;
  constexpr int NR320 = 320, NR960 = 960;  // weight row pads (x64)
  constexpr int DEPTHM1 = 5;

  char* ws = (char*)d_ws;
  size_t off = 0;
  auto alloc = [&](size_t bytes) -> void* {
    void* p = ws + off;
    off += (bytes + 255) & ~(size_t)255;
    return p;
  };

  bf16* fa_bf   = (bf16*)alloc((size_t)A_ * Fp * 2);
  bf16* fb_bf   = (bf16*)alloc((size_t)B_ * Fp * 2);
  bf16* wiA_bf  = (bf16*)alloc((size_t)NR320 * Fp * 2);
  bf16* wiB_bf  = (bf16*)alloc((size_t)NR320 * Fp * 2);
  bf16* wh_bf   = (bf16*)alloc((size_t)DEPTHM1 * NR320 * Hp * 2);
  bf16* wlr_bf  = (bf16*)alloc((size_t)NR320 * H3p * 2);
  bf16* wih_bf  = (bf16*)alloc((size_t)2 * NR960 * Hp * 2);
  bf16* whh_bf  = (bf16*)alloc((size_t)2 * NR960 * Hp * 2);
  bf16* wo_bf   = (bf16*)alloc((size_t)NR320 * H2p * 2);
  float* inA    = (float*)alloc((size_t)A_ * H * 4);   // input_atom
  float* inB    = (float*)alloc((size_t)B_ * H * 4);   // input_bond
  float* mA     = (float*)alloc((size_t)A_ * H * 4);   // message_atom
  float* mB     = (float*)alloc((size_t)B_ * H * 4);   // message_bond
  bf16* tmp_bf  = (bf16*)alloc((size_t)B_ * Hp * 2);
  bf16* cat_bf  = (bf16*)alloc((size_t)A_ * H3p * 2);
  float* node   = (float*)alloc((size_t)A_ * H * 4);
  bf16* x_bf    = (bf16*)alloc((size_t)NT * Hp * 2);
  float* gi0    = (float*)alloc((size_t)NT * H3 * 4);
  float* gi1    = (float*)alloc((size_t)NT * H3 * 4);
  float* gh0    = (float*)alloc((size_t)NM * H3 * 4);
  float* gh1    = (float*)alloc((size_t)NM * H3 * 4);
  float* hf0    = (float*)alloc((size_t)NM * H * 4);
  float* hf1    = (float*)alloc((size_t)NM * H * 4);
  bf16* hb0     = (bf16*)alloc((size_t)NM * Hp * 2);
  bf16* hb1     = (bf16*)alloc((size_t)NM * Hp * 2);
  bf16* full_bf = (bf16*)alloc((size_t)NT * H2p * 2);

  auto cvt = [&](const float* in, int M, int K, bf16* out, int Mo, int Ko) {
    size_t tot = (size_t)Mo * Ko;
    int blocks = (int)((tot + 255) / 256);
    cvt_pad_bf16<<<blocks, 256, 0, stream>>>(in, M, K, out, Mo, Ko);
  };
  auto gemm = [&](const bf16* Amat, int lda, int M, const bf16* Wm, int N, int K,
                  const float* addend, int ldadd, const float* bias,
                  float* C, int ldc, int relu) {
    dim3 grid((M + 255) / 256, (N + 63) / 64);
    size_t shmem = (size_t)64 * (K + 8) * sizeof(bf16);   // W panel in LDS
    gemm_bf16_wmma<<<grid, 256, shmem, stream>>>(Amat, lda, M, Wm, K,
                                                 addend, ldadd, bias, C, ldc, N, K, relu);
  };

  // ---- convert inputs & weights to padded bf16 ----
  cvt(f_atoms, A_, AF, fa_bf, A_, Fp);
  cvt(f_bonds, B_, BF, fb_bf, B_, Fp);
  cvt(Wi_atom, H, AF, wiA_bf, NR320, Fp);
  cvt(Wi_bond, H, BF, wiB_bf, NR320, Fp);
  for (int d = 0; d < DEPTHM1; ++d)
    cvt(Wh + (size_t)d * H * H, H, H, wh_bf + (size_t)d * NR320 * Hp, NR320, Hp);
  cvt(W_lr, H, H3, wlr_bf, NR320, H3p);
  for (int dir = 0; dir < 2; ++dir) {
    cvt(gru_wih + (size_t)dir * H3 * H, H3, H, wih_bf + (size_t)dir * NR960 * Hp, NR960, Hp);
    cvt(gru_whh + (size_t)dir * H3 * H, H3, H, whh_bf + (size_t)dir * NR960 * Hp, NR960, Hp);
  }
  cvt(Wo_w, H, H2, wo_bf, NR320, H2p);

  // ---- input projections ----
  gemm(fa_bf, Fp, A_, wiA_bf, H, Fp, nullptr, 0, nullptr, inA, H, 1);
  gemm(fb_bf, Fp, B_, wiB_bf, H, Fp, nullptr, 0, nullptr, inB, H, 1);
  hipMemcpyAsync(mA, inA, (size_t)A_ * H * 4, hipMemcpyDeviceToDevice, stream);
  hipMemcpyAsync(mB, inB, (size_t)B_ * H * 4, hipMemcpyDeviceToDevice, stream);

  // ---- message passing loop ----
  for (int d = 0; d < DEPTHM1; ++d) {
    atom_agg_k<<<A_, 256, 0, stream>>>(mB, a2b, mA, H);
    bond_tmp_k<<<B_, 256, 0, stream>>>(mA, mB, b2a, b2revb, tmp_bf, H, Hp);
    gemm(tmp_bf, Hp, B_, wh_bf + (size_t)d * NR320 * Hp, H, Hp, inB, H, nullptr, mB, H, 1);
  }

  // ---- readout: node = concat @ W_lr^T ----
  final_concat_k<<<A_, 256, 0, stream>>>(mB, a2b, mA, inA, cat_bf, H, H3p);
  gemm(cat_bf, H3p, A_, wlr_bf, H, H3p, nullptr, 0, nullptr, node, H, 0);

  // ---- GRU setup ----
  msg_x_k<<<NT, 256, 0, stream>>>(node, gru_bias, x_bf, H, Hp);
  h0_k<<<NM, 256, 0, stream>>>(node, hf0, hf1, hb0, hb1, H, Hp, T);
  fill_pad_k<<<(NT * (H2p - H2) + 255) / 256, 256, 0, stream>>>(full_bf, NT, H2p, H2, H2p);

  // gi = x @ wih^T + bih (both directions, one big GEMM each)
  gemm(x_bf, Hp, NT, wih_bf,                      H3, Hp, nullptr, 0, gru_bih,      gi0, H3, 0);
  gemm(x_bf, Hp, NT, wih_bf + (size_t)NR960 * Hp, H3, Hp, nullptr, 0, gru_bih + H3, gi1, H3, 0);

  // ---- recurrent steps ----
  for (int t = 0; t < T; ++t) {
    // forward
    gemm(hb0, Hp, NM, whh_bf, H3, Hp, nullptr, 0, gru_bhh, gh0, H3, 0);
    gru_gate_k<<<NM, 256, 0, stream>>>(gi0, gh0, hf0, hb0, full_bf, t, 0, H, Hp, H2p, T);
    // backward (time-reversed)
    gemm(hb1, Hp, NM, whh_bf + (size_t)NR960 * Hp, H3, Hp, nullptr, 0, gru_bhh + H3, gh1, H3, 0);
    gru_gate_k<<<NM, 256, 0, stream>>>(gi1, gh1, hf1, hb1, full_bf, T - 1 - t, H, H, Hp, H2p, T);
  }

  // ---- output projection: atom_hiddens[1:] ----
  gemm(full_bf, H2p, NT, wo_bf, H, H2p, nullptr, 0, Wo_b, (float*)d_out, H, 1);

  (void)in_sizes; (void)n_in; (void)out_size; (void)ws_size; (void)gru_bias;
}
